// selective_channel_aggregation_Module2_12807592476761
// MI455X (gfx1250) — matrix operand order repaired and checked by reading, not compiled, not run on hardware
//
#include <hip/hip_runtime.h>
#include <hip/hip_bf16.h>

typedef __attribute__((ext_vector_type(16))) _Float16 v16h;
typedef __attribute__((ext_vector_type(8)))  _Float16 v8h;
typedef __attribute__((ext_vector_type(8)))  float    v8f;

#define BB 8
#define CC 512
#define NN 16384
#define QQ 128
#define KSTR 40          // LDS row stride in halfs (32 + 8 pad)
#define KCH  1024        // split-K chunk for the energy GEMM
#define NKC  16          // number of K chunks (16384/1024)

// ---------------------------------------------------------------------------
// fp32 -> fp16 conversion (vectorized x8)
// ---------------------------------------------------------------------------
__global__ __launch_bounds__(256) void cvt_f32_f16_kernel(
    const float* __restrict__ src, _Float16* __restrict__ dst, int n) {
  int i = (blockIdx.x * blockDim.x + threadIdx.x) * 8;
  int stride = gridDim.x * blockDim.x * 8;
  for (; i < n; i += stride) {
    float4 a = *(const float4*)(src + i);
    float4 b = *(const float4*)(src + i + 4);
    v8h h;
    h[0] = (_Float16)a.x; h[1] = (_Float16)a.y;
    h[2] = (_Float16)a.z; h[3] = (_Float16)a.w;
    h[4] = (_Float16)b.x; h[5] = (_Float16)b.y;
    h[6] = (_Float16)b.z; h[7] = (_Float16)b.w;
    *(v8h*)(dst + i) = h;
  }
}

// ---------------------------------------------------------------------------
// fragment helpers (layouts per CDNA5 ISA 7.12.2, wave32)
// ---------------------------------------------------------------------------
__device__ inline v16h cat8(v8h lo, v8h hi) {
  return __builtin_shufflevector(lo, hi, 0, 1, 2, 3, 4, 5, 6, 7,
                                 8, 9, 10, 11, 12, 13, 14, 15);
}

// A 16x32 f16 frag: lane<16 holds K{0..7,16..23}, lane>=16 holds K{8..15,24..31}
__device__ inline v16h ld_frag_a(const _Float16* row_ptr, int hl) {
  const _Float16* p = row_ptr + hl * 8;
  return cat8(*(const v8h*)p, *(const v8h*)(p + 16));
}
// B 32x16 f16 frag (stored transposed in LDS as [n][K]): lane<16 holds K{0..7,16..23}, lane>=16 holds K{8..15,24..31}
__device__ inline v16h ld_frag_b(const _Float16* row_ptr, int hl) {
  const _Float16* p = row_ptr + hl * 8;
  return cat8(*(const v8h*)p, *(const v8h*)(p + 16));
}

// Preload A frag + all 8 B frags, then run the WMMA chain: lets the in-order
// DS returns stagger the s_wait_dscnt counts instead of one wait per WMMA.
__device__ inline void compute_tile(const _Float16* sA, const _Float16* sB,
                                    v8f* acc, int wv, int l15, int hl) {
  v16h af = ld_frag_a(&sA[(wv * 16 + l15) * KSTR], hl);
  v16h bf[8];
#pragma unroll
  for (int j = 0; j < 8; ++j)
    bf[j] = ld_frag_b(&sB[(j * 16 + l15) * KSTR], hl);
#pragma unroll
  for (int j = 0; j < 8; ++j)
    acc[j] = __builtin_amdgcn_wmma_f32_16x16x32_f16(
        false, af, false, bf[j], (short)0, acc[j], false, false);
}

// ---------------------------------------------------------------------------
// GEMM over K=C=512 producing a [128(Q) x 128(n)] tile. Double-buffered LDS,
// register-staged global loads, one barrier per K-step.
// MODE 0:  D = Wq*x       -> BN + leaky-relu -> d_out (q) and q_h (fp16)
// MODE 1:  D = attn'*x    -> d_out += D   (attn' already has gamma folded in)
// ---------------------------------------------------------------------------
template <int MODE>
__global__ __launch_bounds__(256) void gemm_qn_kernel(
    const _Float16* __restrict__ Ab,      // MODE0: Wq_h[128][512]; MODE1: attn_h[B][128][512]
    const _Float16* __restrict__ xh,      // [B][512][16384]
    const float* __restrict__ bns, const float* __restrict__ bnb,
    const float* __restrict__ bnm, const float* __restrict__ bnv,
    float* __restrict__ outp,             // [B][128][16384]
    _Float16* __restrict__ qh)            // [B][128][16384] (MODE0 only)
{
  __shared__ _Float16 sA[2][128 * KSTR];
  __shared__ _Float16 sB[2][128 * KSTR];

  const int b  = blockIdx.y;
  const int n0 = blockIdx.x * 128;
  const int t  = threadIdx.x;
  const int lane = t & 31, wv = t >> 5, hl = lane >> 4, l15 = lane & 15;

  const _Float16* Asrc = (MODE == 0) ? Ab : (Ab + (size_t)b * QQ * CC);
  const _Float16* Bsrc = xh + (size_t)b * CC * NN + n0;

  const int aq = t >> 1, ak = (t & 1) * 16;   // A stage role
  const int bk = t >> 3, bj = (t & 7) * 16;   // B stage role (transpose)

  v8h ra0, ra1, rb0, rb1;                     // in-flight staging registers

  auto load_regs = [&](int kc) {
    const _Float16* pa = Asrc + (size_t)aq * CC + kc + ak;
    ra0 = *(const v8h*)pa;
    ra1 = *(const v8h*)(pa + 8);
    const _Float16* pb = Bsrc + (size_t)(kc + bk) * NN + bj;
    rb0 = *(const v8h*)pb;
    rb1 = *(const v8h*)(pb + 8);
    __builtin_prefetch(pb + (size_t)64 * NN, 0, 1);   // warm L2 two tiles ahead
  };
  auto store_tile = [&](int buf) {
    *(v8h*)&sA[buf][aq * KSTR + ak]     = ra0;
    *(v8h*)&sA[buf][aq * KSTR + ak + 8] = ra1;
#pragma unroll
    for (int i = 0; i < 8; ++i) {               // transpose: LDS[n][k]
      sB[buf][(bj + i) * KSTR + bk]     = rb0[i];
      sB[buf][(bj + 8 + i) * KSTR + bk] = rb1[i];
    }
  };

  v8f acc[8] = {};

  load_regs(0);
  store_tile(0);
  load_regs(32);
  __syncthreads();

  const int ITERS = CC / 32;                    // 16
  for (int it = 0; it < ITERS; ++it) {
    if (it + 1 < ITERS) store_tile((it + 1) & 1);
    if (it + 2 < ITERS) load_regs((it + 2) * 32);
    compute_tile(sA[it & 1], sB[it & 1], acc, wv, l15, hl);
    __syncthreads();
  }

  if (MODE == 0) {
    float mn[8], sd[8], bi[8];
#pragma unroll
    for (int r = 0; r < 8; ++r) {
      int M = wv * 16 + hl * 8 + r;
      mn[r] = bnm[M];
      sd[r] = bns[M] * rsqrtf(bnv[M] + 1e-5f);
      bi[r] = bnb[M];
    }
#pragma unroll
    for (int j = 0; j < 8; ++j) {
      int col = n0 + j * 16 + l15;
#pragma unroll
      for (int r = 0; r < 8; ++r) {
        int M = wv * 16 + hl * 8 + r;
        float v = (acc[j][r] - mn[r]) * sd[r] + bi[r];
        v = (v >= 0.f) ? v : 0.01f * v;          // leaky_relu(0.01)
        size_t idx = ((size_t)(b * QQ + M)) * NN + col;
        outp[idx] = v;
        qh[idx] = (_Float16)v;
      }
    }
  } else {
#pragma unroll
    for (int j = 0; j < 8; ++j) {
      int col = n0 + j * 16 + l15;
#pragma unroll
      for (int r = 0; r < 8; ++r) {
        int M = wv * 16 + hl * 8 + r;
        size_t idx = ((size_t)(b * QQ + M)) * NN + col;
        outp[idx] = outp[idx] + acc[j][r];       // gamma*out_c + q
      }
    }
  }
}

// ---------------------------------------------------------------------------
// energy partials: [128(Q) x 128(c)] tile, split-K chunk of 1024 over K=N.
// part[b][ct][kc][128][128]  (summed deterministically in softmax kernel)
// ---------------------------------------------------------------------------
__global__ __launch_bounds__(256) void gemm_energy_kernel(
    const _Float16* __restrict__ qh,      // [B][128][16384]
    const _Float16* __restrict__ xh,      // [B][512][16384]
    float* __restrict__ part)
{
  __shared__ _Float16 sA[2][128 * KSTR];
  __shared__ _Float16 sB[2][128 * KSTR];

  const int kchunk = blockIdx.x;
  const int ct = blockIdx.y;
  const int b  = blockIdx.z;
  const int c0 = ct * 128;
  const int kbase = kchunk * KCH;

  const int t = threadIdx.x;
  const int lane = t & 31, wv = t >> 5, hl = lane >> 4, l15 = lane & 15;
  const int row = t >> 1, koff = (t & 1) * 16;

  const _Float16* Apnl = qh + ((size_t)(b * QQ + row)) * NN + kbase + koff;
  const _Float16* Bpnl = xh + ((size_t)(b * CC + c0 + row)) * NN + kbase + koff;

  v8h ra0, ra1, rb0, rb1;

  auto load_regs = [&](int kk) {
    const _Float16* pa = Apnl + kk;
    const _Float16* pb = Bpnl + kk;
    ra0 = *(const v8h*)pa;
    ra1 = *(const v8h*)(pa + 8);
    rb0 = *(const v8h*)pb;
    rb1 = *(const v8h*)(pb + 8);
    __builtin_prefetch(pa + 64, 0, 1);
    __builtin_prefetch(pb + 64, 0, 1);
  };
  auto store_tile = [&](int buf) {
    *(v8h*)&sA[buf][row * KSTR + koff]     = ra0;
    *(v8h*)&sA[buf][row * KSTR + koff + 8] = ra1;
    *(v8h*)&sB[buf][row * KSTR + koff]     = rb0;
    *(v8h*)&sB[buf][row * KSTR + koff + 8] = rb1;
  };

  v8f acc[8] = {};

  load_regs(0);
  store_tile(0);
  load_regs(32);
  __syncthreads();

  const int ITERS = KCH / 32;                   // 32
  for (int it = 0; it < ITERS; ++it) {
    if (it + 1 < ITERS) store_tile((it + 1) & 1);
    if (it + 2 < ITERS) load_regs((it + 2) * 32);
    compute_tile(sA[it & 1], sB[it & 1], acc, wv, l15, hl);
    __syncthreads();
  }

#pragma unroll
  for (int j = 0; j < 8; ++j) {
#pragma unroll
    for (int r = 0; r < 8; ++r) {
      int qrow = wv * 16 + hl * 8 + r;
      int ccol = j * 16 + l15;
      size_t idx = ((((size_t)(b * 4 + ct) * NKC) + kchunk) * 128 + qrow) * 128 + ccol;
      part[idx] = acc[j][r];
    }
  }
}

// ---------------------------------------------------------------------------
// deterministic split-K reduce + softmax(rowmax - e) + gamma fold -> attn_h f16
// one wave per (b,q) row of 512
// ---------------------------------------------------------------------------
__global__ __launch_bounds__(256) void reduce_softmax_kernel(
    const float* __restrict__ part, const float* __restrict__ gamma,
    _Float16* __restrict__ attnh)
{
  const int gw = blockIdx.x * 8 + (threadIdx.x >> 5);
  const int lane = threadIdx.x & 31;
  const int b = gw >> 7;
  const int q = gw & 127;

  float e[16];
#pragma unroll
  for (int i = 0; i < 16; ++i) {
    int c = lane + 32 * i;
    int ct = c >> 7, cc = c & 127;
    float s = 0.f;
#pragma unroll
    for (int kc = 0; kc < NKC; ++kc)
      s += part[((((size_t)(b * 4 + ct) * NKC) + kc) * 128 + q) * 128 + cc];
    e[i] = s;
  }
  // softmax(rowmax - e) == exp(rowmin - e) / sum
  float mn = e[0];
#pragma unroll
  for (int i = 1; i < 16; ++i) mn = fminf(mn, e[i]);
  for (int m = 16; m; m >>= 1) mn = fminf(mn, __shfl_xor(mn, m, 32));

  float p[16];
  float sum = 0.f;
#pragma unroll
  for (int i = 0; i < 16; ++i) { p[i] = __expf(mn - e[i]); sum += p[i]; }
  for (int m = 16; m; m >>= 1) sum += __shfl_xor(sum, m, 32);

  const float g = gamma[0] / sum;   // fold gamma into attention
#pragma unroll
  for (int i = 0; i < 16; ++i) {
    int c = lane + 32 * i;
    attnh[((size_t)(b * QQ) + q) * CC + c] = (_Float16)(p[i] * g);
  }
}

// ---------------------------------------------------------------------------
// launcher
// ---------------------------------------------------------------------------
extern "C" void kernel_launch(void* const* d_in, const int* in_sizes, int n_in,
                              void* d_out, int out_size, void* d_ws, size_t ws_size,
                              hipStream_t stream) {
  const float* x   = (const float*)d_in[0];
  const float* Wq  = (const float*)d_in[1];
  const float* bns = (const float*)d_in[2];
  const float* bnb = (const float*)d_in[3];
  const float* bnm = (const float*)d_in[4];
  const float* bnv = (const float*)d_in[5];
  const float* gm  = (const float*)d_in[6];
  float* outp = (float*)d_out;

  char* ws = (char*)d_ws;
  _Float16* xh    = (_Float16*)(ws);                        // 128 MB
  _Float16* qh    = (_Float16*)(ws + 134217728ull);         //  32 MB
  _Float16* Wqh   = (_Float16*)(ws + 167772160ull);         // 128 KB
  _Float16* attnh = (_Float16*)(ws + 167903232ull);         //   1 MB
  float*    part  = (float*)   (ws + 168951808ull);         //  32 MB

  // 1) fp32 -> fp16 (x fits in L2 afterwards: 128 MB < 192 MB)
  cvt_f32_f16_kernel<<<32768, 256, 0, stream>>>(x, xh, BB * CC * NN);
  cvt_f32_f16_kernel<<<32, 256, 0, stream>>>(Wq, Wqh, QQ * CC);

  // 2) q = leaky(BN(Wq*x)) -> d_out (f32) and q_h (f16)
  gemm_qn_kernel<0><<<dim3(128, BB), 256, 0, stream>>>(
      Wqh, xh, bns, bnb, bnm, bnv, outp, qh);

  // 3) energy partials (split-K, deterministic)
  gemm_energy_kernel<<<dim3(NKC, 4, BB), 256, 0, stream>>>(qh, xh, part);

  // 4) reduce + softmax + gamma fold -> attn_h (f16)
  reduce_softmax_kernel<<<128, 256, 0, stream>>>(part, gm, attnh);

  // 5) d_out = attn'*x + q
  gemm_qn_kernel<1><<<dim3(128, BB), 256, 0, stream>>>(
      attnh, xh, bns, bnb, bnm, bnv, outp, (_Float16*)nullptr);
}
